// PatchGAT_34514357191322
// MI455X (gfx1250) — matrix operand order repaired and checked by reading, not compiled, not run on hardware
//
#include <hip/hip_runtime.h>

// ---------------------------------------------------------------------------
// Types / helpers
// ---------------------------------------------------------------------------
typedef __attribute__((ext_vector_type(16))) _Float16 v16h;
typedef __attribute__((ext_vector_type(8)))  float    v8f;

union Frag16 { v16h h; float4 f[2]; };

#define NEG_SLOPE 0.2f

__device__ __forceinline__ void atomicMaxF(float* addr, float v) {
  // order-preserving int punning (valid for mixed-sign floats)
  if (v >= 0.0f) atomicMax((int*)addr, __float_as_int(v));
  else           atomicMin((unsigned int*)addr, __float_as_uint(v));
}

// ---------------------------------------------------------------------------
// Generic fill / convert kernels
// ---------------------------------------------------------------------------
__global__ void k_fill(float* __restrict__ p, float v, long long n) {
  long long i = (long long)blockIdx.x * blockDim.x + threadIdx.x;
  if (i < n) p[i] = v;
}

__global__ void k_cvt_f16(const float* __restrict__ in, _Float16* __restrict__ out,
                          long long n) {
  long long i = (long long)blockIdx.x * blockDim.x + threadIdx.x;
  if (i < n) out[i] = (_Float16)in[i];
}

// W [K, Nm] row-major f32  ->  WT [Nm, K] row-major f16
__global__ void k_cvt_tr_f16(const float* __restrict__ W, _Float16* __restrict__ WT,
                             int K, int Nm) {
  int i = blockIdx.x * blockDim.x + threadIdx.x;
  if (i >= K * Nm) return;
  int k = i / Nm;
  int n = i - k * Nm;
  WT[(long long)n * K + k] = (_Float16)W[i];
}

// ---------------------------------------------------------------------------
// WMMA GEMM:  C[M,Nm] (f32) = A[M,K] (f16, row-major) x BT[Nm,K]^T (f16)
// One wave computes a 16x64 strip of C (4 wmma accumulators, shared A frag).
// Optional per-column bias + ReLU fused into the epilogue.
// M % 16 == 0, Nm % 64 == 0, K % 32 == 0 (true for all calls here).
// ---------------------------------------------------------------------------
__global__ __launch_bounds__(256) void k_gemm_wmma_f16(
    const _Float16* __restrict__ A, const _Float16* __restrict__ BT,
    float* __restrict__ C, const float* __restrict__ bias,
    int M, int Nm, int K, int fuse_relu) {
  const int lane  = threadIdx.x & 31;
  const int wave  = blockIdx.x * (blockDim.x >> 5) + (threadIdx.x >> 5);
  const int tilesN = Nm >> 6;                       // 64-wide N supertiles
  const int totalWaves = (M >> 4) * tilesN;
  if (wave >= totalWaves) return;                   // wave-uniform: EXEC stays all-1s

  const int tm   = wave / tilesN;
  const int tn   = (wave - tm * tilesN) << 6;
  const int l15  = lane & 15;
  const int hi   = (lane >> 4) & 1;
  const int mrow = (tm << 4) + l15;

  v8f acc[4] = {};

  for (int k = 0; k < K; k += 32) {
    // A fragment: lane holds row mrow; K chunks [kb, kb+8) and [kb+16, kb+24)
    const int kbA = k + (hi << 3);
    Frag16 a;
    const _Float16* ap = A + (long long)mrow * K + kbA;
    a.f[0] = *reinterpret_cast<const float4*>(ap);
    a.f[1] = *reinterpret_cast<const float4*>(ap + 16);

    // B fragment: lane holds column (tn + s*16 + l15); K chunks [kb, kb+8) and [kb+16, kb+24)
    const int kbB = k + (hi << 3);
#pragma unroll
    for (int s = 0; s < 4; ++s) {
      const int ncol = tn + (s << 4) + l15;
      Frag16 b;
      const _Float16* bp = BT + (long long)ncol * K + kbB;
      b.f[0] = *reinterpret_cast<const float4*>(bp);
      b.f[1] = *reinterpret_cast<const float4*>(bp + 16);
      acc[s] = __builtin_amdgcn_wmma_f32_16x16x32_f16(
          /*neg_a=*/false, a.h, /*neg_b=*/false, b.h,
          /*c_mod=*/(short)0, acc[s], /*reuse_a=*/false, /*reuse_b=*/false);
    }
  }

  // Epilogue: lanes 0-15 hold M=tm*16+r, lanes 16-31 hold M=tm*16+8+r
  const int rbase = (tm << 4) + (hi << 3);
#pragma unroll
  for (int s = 0; s < 4; ++s) {
    const int col = tn + (s << 4) + l15;
    const float bv = bias ? bias[col] : 0.0f;
#pragma unroll
    for (int r = 0; r < 8; ++r) {
      float v = acc[s][r] + bv;
      if (fuse_relu) v = v > 0.0f ? v : 0.0f;
      C[(long long)(rbase + r) * Nm + col] = v;
    }
  }
}

// ---------------------------------------------------------------------------
// Attention scores: el/er [N,H] = sum_d feat[n,h,d] * attn[h,d]
// ---------------------------------------------------------------------------
__global__ void k_attn_scores(const float* __restrict__ feat,
                              const float* __restrict__ al,
                              const float* __restrict__ ar,
                              float* __restrict__ el, float* __restrict__ er,
                              int N, int H, int D) {
  int i = blockIdx.x * blockDim.x + threadIdx.x;
  if (i >= N * H) return;
  int n = i / H, h = i - n * H;
  const float* f = feat + (long long)n * H * D + h * D;
  const float* a = al + h * D;
  const float* b = ar + h * D;
  float sl = 0.0f, sr = 0.0f;
  for (int d = 0; d < D; ++d) {
    float fv = f[d];
    sl += fv * a[d];
    sr += fv * b[d];
  }
  el[i] = sl;
  er[i] = sr;
}

// ---------------------------------------------------------------------------
// Edge softmax, pass 1: running max per (dst, head)
// ---------------------------------------------------------------------------
__global__ void k_edge_max(const int* __restrict__ src, const int* __restrict__ dst,
                           const float* __restrict__ el, const float* __restrict__ er,
                           float* __restrict__ m, int E, int H) {
  int e = blockIdx.x * blockDim.x + threadIdx.x;
  if (e >= E) return;
  int s0 = src[e], d0 = dst[e];
  for (int h = 0; h < H; ++h) {
    float v = el[s0 * H + h] + er[d0 * H + h];
    v = v > 0.0f ? v : NEG_SLOPE * v;               // leaky_relu
    atomicMaxF(&m[d0 * H + h], v);
  }
}

// ---------------------------------------------------------------------------
// Edge softmax, pass 2: ex = exp(e - m[dst]);  s[dst] += ex
// ---------------------------------------------------------------------------
__global__ void k_edge_exp(const int* __restrict__ src, const int* __restrict__ dst,
                           const float* __restrict__ el, const float* __restrict__ er,
                           const float* __restrict__ m, float* __restrict__ ssum,
                           float* __restrict__ ex, int E, int H) {
  int e = blockIdx.x * blockDim.x + threadIdx.x;
  if (e >= E) return;
  int s0 = src[e], d0 = dst[e];
  for (int h = 0; h < H; ++h) {
    float v = el[s0 * H + h] + er[d0 * H + h];
    v = v > 0.0f ? v : NEG_SLOPE * v;
    float xv = __expf(v - m[d0 * H + h]);
    ex[(long long)e * H + h] = xv;
    atomicAdd(&ssum[d0 * H + h], xv);
  }
}

// ---------------------------------------------------------------------------
// Scatter aggregate: rst[dst,h,:] += (ex/s[dst,h]) * feat[src,h,:]
// One block per edge; blockDim = H*D/4; each thread handles 4 contiguous dims.
// feat is L2-resident (40 MB << 192 MB), so gathers and atomics hit L2.
// ---------------------------------------------------------------------------
__global__ void k_edge_agg(const int* __restrict__ src, const int* __restrict__ dst,
                           const float* __restrict__ ex, const float* __restrict__ ssum,
                           const float* __restrict__ feat, float* __restrict__ rst,
                           int H, int D) {
  int e = blockIdx.x;
  int j = threadIdx.x * 4;                          // dim within flattened H*D
  int s0 = src[e], d0 = dst[e];
  int h = j / D;
  float alpha = ex[(long long)e * H + h] / ssum[d0 * H + h];
  const float4 f = *reinterpret_cast<const float4*>(
      feat + (long long)s0 * H * D + j);
  float* o = rst + (long long)d0 * H * D + j;
  atomicAdd(o + 0, alpha * f.x);
  atomicAdd(o + 1, alpha * f.y);
  atomicAdd(o + 2, alpha * f.z);
  atomicAdd(o + 3, alpha * f.w);
}

// ---------------------------------------------------------------------------
// GAT epilogue: rst = relu(rst + res + bias)
// ---------------------------------------------------------------------------
__global__ void k_bias_relu(float* __restrict__ rst, const float* __restrict__ res,
                            const float* __restrict__ bias, int HD, long long total) {
  long long i = (long long)blockIdx.x * blockDim.x + threadIdx.x;
  if (i >= total) return;
  float v = rst[i] + res[i] + bias[i % HD];
  rst[i] = v > 0.0f ? v : 0.0f;
}

// ---------------------------------------------------------------------------
// Final: relu(mean over heads) then max over nodes -> out[D]
// blockDim = D; each block scans a 256-node chunk; one atomic per (block, d).
// All values >= 0 after relu, so non-negative int-punned atomicMax is exact.
// ---------------------------------------------------------------------------
__global__ void k_pool_max(const float* __restrict__ rst, float* __restrict__ out,
                           int N, int H, int D) {
  int d  = threadIdx.x;
  int n0 = blockIdx.x * 256;
  int n1 = n0 + 256; if (n1 > N) n1 = N;
  float mx = 0.0f;
  float invH = 1.0f / (float)H;
  for (int n = n0; n < n1; ++n) {
    const float* r = rst + (long long)n * H * D + d;
    float s = 0.0f;
    for (int h = 0; h < H; ++h) s += r[h * D];
    s *= invH;
    s = s > 0.0f ? s : 0.0f;
    mx = fmaxf(mx, s);
  }
  atomicMax((int*)&out[d], __float_as_int(mx));
}

// ---------------------------------------------------------------------------
// Host-side layer drivers (all enqueues on `stream`; stream order gives deps)
// ---------------------------------------------------------------------------
static void launch_gemm(hipStream_t stream, const _Float16* A, const _Float16* BT,
                        float* C, const float* bias, int M, int Nm, int K, int relu) {
  int totalWaves = (M >> 4) * (Nm >> 6);
  int blocks = (totalWaves + 7) / 8;                // 8 waves / 256-thread block
  k_gemm_wmma_f16<<<blocks, 256, 0, stream>>>(A, BT, C, bias, M, Nm, K, relu);
}

static void gat_layer(hipStream_t stream, const float* xin, int Kin,
                      const int* src, const int* dst, int N, int E, int H, int D,
                      const float* W, const float* rW, const float* al,
                      const float* ar, const float* bias,
                      _Float16* a16, _Float16* w16,
                      float* feat, float* res, float* el, float* er,
                      float* m, float* s, float* ex, float* rst) {
  const int HD = H * D;
  long long nx = (long long)N * Kin;
  k_cvt_f16<<<(int)((nx + 255) / 256), 256, 0, stream>>>(xin, a16, nx);

  k_cvt_tr_f16<<<(Kin * HD + 255) / 256, 256, 0, stream>>>(W, w16, Kin, HD);
  launch_gemm(stream, a16, w16, feat, nullptr, N, HD, Kin, 0);

  k_cvt_tr_f16<<<(Kin * HD + 255) / 256, 256, 0, stream>>>(rW, w16, Kin, HD);
  launch_gemm(stream, a16, w16, res, nullptr, N, HD, Kin, 0);

  k_attn_scores<<<(N * H + 255) / 256, 256, 0, stream>>>(feat, al, ar, el, er, N, H, D);

  long long nh = (long long)N * H;
  long long nhd = (long long)N * HD;
  k_fill<<<(int)((nh + 255) / 256), 256, 0, stream>>>(m, -__builtin_huge_valf(), nh);
  k_fill<<<(int)((nh + 255) / 256), 256, 0, stream>>>(s, 0.0f, nh);
  k_fill<<<(int)((nhd + 255) / 256), 256, 0, stream>>>(rst, 0.0f, nhd);

  k_edge_max<<<(E + 255) / 256, 256, 0, stream>>>(src, dst, el, er, m, E, H);
  k_edge_exp<<<(E + 255) / 256, 256, 0, stream>>>(src, dst, el, er, m, s, ex, E, H);
  k_edge_agg<<<E, HD / 4, 0, stream>>>(src, dst, ex, s, feat, rst, H, D);

  k_bias_relu<<<(int)((nhd + 255) / 256), 256, 0, stream>>>(rst, res, bias, HD, nhd);
}

static void dense_layer(hipStream_t stream, const float* xin, int Kin,
                        const float* DW, const float* Db, int Nout, float* out,
                        _Float16* a16, _Float16* w16, int N) {
  long long nx = (long long)N * Kin;
  k_cvt_f16<<<(int)((nx + 255) / 256), 256, 0, stream>>>(xin, a16, nx);
  k_cvt_tr_f16<<<(Kin * Nout + 255) / 256, 256, 0, stream>>>(DW, w16, Kin, Nout);
  launch_gemm(stream, a16, w16, out, Db, N, Nout, Kin, 1);
}

// ---------------------------------------------------------------------------
// Entry point
// ---------------------------------------------------------------------------
extern "C" void kernel_launch(void* const* d_in, const int* in_sizes, int n_in,
                              void* d_out, int out_size, void* d_ws, size_t ws_size,
                              hipStream_t stream) {
  const float* n_feat = (const float*)d_in[0];
  const int*   src    = (const int*)d_in[1];
  const int*   dst    = (const int*)d_in[2];
  const float* W0  = (const float*)d_in[3];
  const float* al0 = (const float*)d_in[4];
  const float* ar0 = (const float*)d_in[5];
  const float* b0  = (const float*)d_in[6];
  const float* rW0 = (const float*)d_in[7];
  const float* DW0 = (const float*)d_in[8];
  const float* Db0 = (const float*)d_in[9];
  const float* W1  = (const float*)d_in[10];
  const float* al1 = (const float*)d_in[11];
  const float* ar1 = (const float*)d_in[12];
  const float* b1  = (const float*)d_in[13];
  const float* rW1 = (const float*)d_in[14];
  const float* DW1 = (const float*)d_in[15];
  const float* Db1 = (const float*)d_in[16];
  const float* W2  = (const float*)d_in[17];
  const float* al2 = (const float*)d_in[18];
  const float* ar2 = (const float*)d_in[19];
  const float* b2  = (const float*)d_in[20];
  const float* rW2 = (const float*)d_in[21];

  const int HD   = in_sizes[4];            // H*HID = 512
  const int F_IN = in_sizes[3] / HD;       // 128
  const int N    = in_sizes[0] / F_IN;     // 20000
  const int E    = in_sizes[1];            // 320000
  const int HID  = in_sizes[9];            // 128
  const int H    = HD / HID;               // 4
  const int OUT  = in_sizes[18] / H;       // 128

  // ---- carve workspace (256B aligned) ----
  char* wp = (char*)d_ws;
  auto carve = [&](size_t bytes) -> void* {
    void* p = (void*)wp;
    wp += (bytes + 255) & ~(size_t)255;
    return p;
  };
  float*    feat = (float*)carve((size_t)N * HD * 4);
  float*    res  = (float*)carve((size_t)N * HD * 4);
  float*    rst  = (float*)carve((size_t)N * HD * 4);
  float*    hbuf = (float*)carve((size_t)N * HID * 4);
  _Float16* a16  = (_Float16*)carve((size_t)N * HD * 2);
  _Float16* w16  = (_Float16*)carve((size_t)HD * HD * 2);
  float*    el   = (float*)carve((size_t)N * H * 4);
  float*    er   = (float*)carve((size_t)N * H * 4);
  float*    mbuf = (float*)carve((size_t)N * H * 4);
  float*    sbuf = (float*)carve((size_t)N * H * 4);
  float*    ex   = (float*)carve((size_t)E * H * 4);
  (void)ws_size; (void)n_in; (void)out_size;

  // ---- layer 0: GAT(F_IN -> H x HID) + dense ----
  gat_layer(stream, n_feat, F_IN, src, dst, N, E, H, HID,
            W0, rW0, al0, ar0, b0, a16, w16,
            feat, res, el, er, mbuf, sbuf, ex, rst);
  dense_layer(stream, rst, HD, DW0, Db0, HID, hbuf, a16, w16, N);

  // ---- layer 1: GAT(HID -> H x HID) + dense ----
  gat_layer(stream, hbuf, HID, src, dst, N, E, H, HID,
            W1, rW1, al1, ar1, b1, a16, w16,
            feat, res, el, er, mbuf, sbuf, ex, rst);
  dense_layer(stream, rst, HD, DW1, Db1, HID, hbuf, a16, w16, N);

  // ---- layer 2: GAT(HID -> H x OUT), head-mean + relu + graph max ----
  gat_layer(stream, hbuf, HID, src, dst, N, E, H, OUT,
            W2, rW2, al2, ar2, b2, a16, w16,
            feat, res, el, er, mbuf, sbuf, ex, rst);

  float* out = (float*)d_out;
  k_fill<<<1, 256, 0, stream>>>(out, 0.0f, OUT);
  k_pool_max<<<(N + 255) / 256, OUT, 0, stream>>>(rst, out, N, H, OUT);
}
